// HypergraphConvLayer_40922448396916
// MI455X (gfx1250) — matrix operand order repaired and checked by reading, not compiled, not run on hardware
//
#include <hip/hip_runtime.h>
#include <hip/hip_bf16.h>

// ---------------------------------------------------------------------------
// HypergraphConv layer for MI455X (gfx1250, wave32, WMMA).
// Bandwidth-bound on 3 full reads of the 1 GB incidence matrix H (~130us at
// 23.3 TB/s); all GEMMs run v_wmma_f32_16x16x32_bf16 with fp32->bf16 fused
// into LDS staging (native cvt), double-buffered so the H stream overlaps the
// WMMAs, and bf16 tiles staged with GLOBAL_LOAD_ASYNC_TO_LDS_B128.
// ---------------------------------------------------------------------------

typedef __attribute__((ext_vector_type(16))) __bf16 bf16x16;
typedef __attribute__((ext_vector_type(8)))  float  f32x8;
typedef int v4i __attribute__((vector_size(4 * sizeof(int))));
typedef __attribute__((address_space(3))) v4i lds_v4i;

union FragBF16 { bf16x16 v; uint4 q[2]; };

#define KC   32            // K-chunk per WMMA pass
#define LDT  40            // padded LDS row stride (bf16 elems): 80B, 16B-aligned
#define SPLITK 8

#ifdef __has_builtin
#if __has_builtin(__builtin_amdgcn_global_load_async_to_lds_b128)
#define HAVE_ASYNC_LDS 1
#endif
#endif
#ifndef HAVE_ASYNC_LDS
#define HAVE_ASYNC_LDS 0
#endif

// Native fp32->bf16 convert (RNE); clang lowers this to the hw cvt on gfx1250
// (pairs fuse into packed cvt), vs ~4 VALU ops for a manual bit sequence.
__device__ __forceinline__ __bf16 to_bf16(float f) { return (__bf16)f; }

// 16B global->LDS copy: async (ASYNCcnt-tracked, bypasses VGPRs) if the
// toolchain exposes the gfx1250 builtin, else plain VGPR bounce.
__device__ __forceinline__ void copy_b128_to_lds(void* lds_dst, const void* gsrc) {
#if HAVE_ASYNC_LDS
  __builtin_amdgcn_global_load_async_to_lds_b128(
      (v4i*)gsrc, (lds_v4i*)lds_dst, 0, 0);
#else
  *(uint4*)lds_dst = *(const uint4*)gsrc;
#endif
}

__device__ __forceinline__ void wait_async0() {
#if HAVE_ASYNC_LDS
  asm volatile("s_wait_asynccnt 0x0" ::: "memory");
#endif
}

// A fragment: 16x32 bf16, row-major tile in LDS with stride LDT.
// Lane m=lane&15 holds K = [kh, kh+8) in q[0] and [16+kh, +8) in q[1],
// kh=(lane>>4)*8.  (ISA 7.12.2, 16-bit A-matrix 16x32.)
__device__ __forceinline__ FragBF16 load_frag_a(const __bf16* tile, int lane) {
  FragBF16 f;
  int m  = lane & 15;
  int kh = (lane >> 4) * 8;
  const __bf16* row = tile + m * LDT;
  f.q[0] = *(const uint4*)(row + kh);
  f.q[1] = *(const uint4*)(row + 16 + kh);
  return f;
}

// B fragment: 32x16 bf16 stored TRANSPOSED in LDS (16 rows of n, stride LDT).
// Lane n=lane&15 holds K = [kh, kh+8) in q[0] and [16+kh, +8) in q[1], kh=(lane>>4)*8.
__device__ __forceinline__ FragBF16 load_frag_b(const __bf16* ttile, int lane) {
  FragBF16 f;
  int n  = lane & 15;
  int kh = (lane >> 4) * 8;
  const __bf16* row = ttile + n * LDT;
  f.q[0] = *(const uint4*)(row + kh);
  f.q[1] = *(const uint4*)(row + 16 + kh);
  return f;
}

__device__ __forceinline__ f32x8 wmma_bf16(FragBF16 a, FragBF16 b, f32x8 c) {
  return __builtin_amdgcn_wmma_f32_16x16x32_bf16(
      false, a.v, false, b.v, (short)0, c, false, false);
}

// ---------------------------------------------------------------------------
// 0) zero d_e (it is accumulated with atomics)
// ---------------------------------------------------------------------------
__global__ void zero_f32_kernel(float* p, int n) {
  int i = blockIdx.x * blockDim.x + threadIdx.x;
  if (i < n) p[i] = 0.0f;
}

// ---------------------------------------------------------------------------
// 1) degrees: one coalesced pass over H computes row sums (d_v) exactly and
//    per-thread column partials folded into d_e with one atomicAdd per col.
// ---------------------------------------------------------------------------
__global__ __launch_bounds__(256)
void degrees_kernel(const float* __restrict__ H, float* __restrict__ d_v,
                    float* __restrict__ d_e, int E) {
  __shared__ float red[256];
  const int tid = threadIdx.x;
  const int rowBase = blockIdx.x * 32;
  const int CH = E >> 8;                 // 32 for E=8192
  float colAcc[32];
#pragma unroll
  for (int c = 0; c < 32; ++c) colAcc[c] = 0.0f;

  for (int r = 0; r < 32; ++r) {
    const float* row = H + (size_t)(rowBase + r) * E;
    float rs = 0.0f;
    for (int c = 0; c < CH; ++c) {
      float v = row[c * 256 + tid];
      rs += v;
      colAcc[c] += v;
    }
    red[tid] = rs; __syncthreads();
    for (int s = 128; s > 0; s >>= 1) {
      if (tid < s) red[tid] += red[tid + s];
      __syncthreads();
    }
    if (tid == 0) d_v[rowBase + r] = red[0];
    __syncthreads();
  }
  for (int c = 0; c < CH; ++c) atomicAdd(&d_e[c * 256 + tid], colAcc[c]);
}

// ---------------------------------------------------------------------------
// 2) inverse degree factors
// ---------------------------------------------------------------------------
__global__ void inv_deg_kernel(const float* d_v, const float* d_e,
                               float* inv_v, float* inv_e, int N, int E) {
  int i = blockIdx.x * blockDim.x + threadIdx.x;
  if (i < N) inv_v[i] = 1.0f / (sqrtf(d_v[i]) + 1e-8f);
  if (i < E) inv_e[i] = 1.0f / (d_e[i] + 1e-8f);
}

// ---------------------------------------------------------------------------
// 3) xsT[d][n] = bf16(x[n][d] * inv_v[n])  (LDS transpose, coalesced both ways)
// ---------------------------------------------------------------------------
__global__ __launch_bounds__(256)
void scale_x_kernel(const float* __restrict__ x, const float* __restrict__ inv_v,
                    __bf16* __restrict__ xsT, int N) {
  __shared__ __bf16 T[128][72];
  const int tid = threadIdx.x;
  const int nb = blockIdx.x * 64;
  const int n  = tid >> 2;            // 0..63
  const int dg = (tid & 3) * 32;      // 0,32,64,96
  float iv = inv_v[nb + n];
  const float4* src = (const float4*)(x + (size_t)(nb + n) * 128 + dg);
#pragma unroll
  for (int q = 0; q < 8; ++q) {
    float4 f = src[q];
    T[dg + q * 4 + 0][n] = to_bf16(f.x * iv);
    T[dg + q * 4 + 1][n] = to_bf16(f.y * iv);
    T[dg + q * 4 + 2][n] = to_bf16(f.z * iv);
    T[dg + q * 4 + 3][n] = to_bf16(f.w * iv);
  }
  __syncthreads();
#pragma unroll
  for (int w = 0; w < 32; ++w) {
    int idx = w * 256 + tid;
    int d = idx >> 6, nn = idx & 63;
    xsT[(size_t)d * N + nb + nn] = T[d][nn];
  }
}

// ---------------------------------------------------------------------------
// 4) GEMM1: m1part[split] = Ht(block) @ xs   (split-K x8, deterministic)
//    Double-buffered.  A = H fp32 pipelined through VGPRs as a 4x4 register
//    block transpose (coalesced global reads, ds_store_b64 LDS writes, native
//    cvt).  B = xsT bf16 via async LDS loads.
// ---------------------------------------------------------------------------
__global__ __launch_bounds__(256)
void gemm1_kernel(const float* __restrict__ H, const __bf16* __restrict__ xsT,
                  float* __restrict__ m1part, int N, int E, int kPerSplit) {
  __shared__ __bf16 At[2][128 * LDT];   // Ht tile: [e 0..127][n 0..31]
  __shared__ __bf16 Bt[2][128 * LDT];   // xsT tile: [d 0..127][n 0..31]
  const int eb    = blockIdx.x * 128;
  const int k0    = blockIdx.y * kPerSplit;
  const int tid   = threadIdx.x;
  const int lane  = tid & 31;
  const int wave  = tid >> 5;
  const int wr    = wave & 3;        // e-rows  [wr*32, +32)
  const int wc    = wave >> 2;       // d-cols  [wc*64, +64)
  const int nk    = kPerSplit / KC;

  const int ae0 = (tid & 31) * 4;    // A: e offset (0..124), fastest -> coalesced
  const int an0 = (tid >> 5) * 4;    // A: n offset (0..28)
  const int bd    = tid >> 1;        // B: row d (0..127)
  const int bhalf = (tid & 1) * 16;  // B: n half

  f32x8 zero = {0.f,0.f,0.f,0.f,0.f,0.f,0.f,0.f};
  f32x8 acc[2][4];
#pragma unroll
  for (int i = 0; i < 2; ++i)
#pragma unroll
    for (int j = 0; j < 4; ++j) acc[i][j] = zero;

  float4 fa0, fa1, fa2, fa3;         // in-flight A fp32 4x4 block (rows n0..n0+3)

  auto issueA = [&](int kk) {
    const float* base = H + (size_t)(k0 + kk * KC + an0) * E + eb + ae0;
    fa0 = *(const float4*)(base);
    fa1 = *(const float4*)(base + E);
    fa2 = *(const float4*)(base + 2 * (size_t)E);
    fa3 = *(const float4*)(base + 3 * (size_t)E);
  };
  auto storeA = [&](int buf) {       // 4x4 register transpose, b64 LDS stores
    float c0[4] = {fa0.x, fa0.y, fa0.z, fa0.w};
    float c1[4] = {fa1.x, fa1.y, fa1.z, fa1.w};
    float c2[4] = {fa2.x, fa2.y, fa2.z, fa2.w};
    float c3[4] = {fa3.x, fa3.y, fa3.z, fa3.w};
#pragma unroll
    for (int j = 0; j < 4; ++j) {
      __bf16 t[4] __attribute__((aligned(8)));
      t[0] = to_bf16(c0[j]); t[1] = to_bf16(c1[j]);
      t[2] = to_bf16(c2[j]); t[3] = to_bf16(c3[j]);
      *(uint2*)&At[buf][(ae0 + j) * LDT + an0] = *(const uint2*)t;
    }
  };
  auto issueB = [&](int kk, int buf) {
    const __bf16* g = xsT + (size_t)bd * N + k0 + kk * KC + bhalf;
    copy_b128_to_lds(&Bt[buf][bd * LDT + bhalf],     g);
    copy_b128_to_lds(&Bt[buf][bd * LDT + bhalf + 8], g + 8);
  };

  issueA(0); issueB(0, 0); storeA(0);
  for (int kk = 0; kk < nk; ++kk) {
    const int cur = kk & 1;
    wait_async0();                   // async writes into buf[cur] done
    __syncthreads();                 // + ds stores of A[cur] visible
    if (kk + 1 < nk) { issueA(kk + 1); issueB(kk + 1, 1 - cur); }

    FragBF16 a[2], b[4];
#pragma unroll
    for (int i = 0; i < 2; ++i) a[i] = load_frag_a(&At[cur][(wr * 32 + i * 16) * LDT], lane);
#pragma unroll
    for (int j = 0; j < 4; ++j) b[j] = load_frag_b(&Bt[cur][(wc * 64 + j * 16) * LDT], lane);
#pragma unroll
    for (int i = 0; i < 2; ++i)
#pragma unroll
      for (int j = 0; j < 4; ++j) acc[i][j] = wmma_bf16(a[i], b[j], acc[i][j]);
    __syncthreads();                 // everyone done reading buf[cur]
    if (kk + 1 < nk) storeA(1 - cur);
  }

  // store fp32 partials: row = v + (lane>>4)*8, col = lane&15 (ISA C layout)
  const size_t pbase = (size_t)blockIdx.y * E * 128;
  const int rl = (lane >> 4) * 8;
  const int cl = lane & 15;
#pragma unroll
  for (int i = 0; i < 2; ++i) {
    int ebase = eb + wr * 32 + i * 16 + rl;
#pragma unroll
    for (int j = 0; j < 4; ++j) {
      int d = wc * 64 + j * 16 + cl;
#pragma unroll
      for (int v = 0; v < 8; ++v)
        m1part[pbase + (size_t)(ebase + v) * 128 + d] = acc[i][j][v];
    }
  }
}

// ---------------------------------------------------------------------------
// 5) reduce split-K partials, apply de^-1, store m1 transposed (bf16)
// ---------------------------------------------------------------------------
__global__ void reduce_m1_kernel(const float* __restrict__ m1part,
                                 const float* __restrict__ inv_e,
                                 __bf16* __restrict__ m1T, int E) {
  int idx = blockIdx.x * 256 + threadIdx.x;   // e*128 + d
  int e = idx >> 7, d = idx & 127;
  size_t stride = (size_t)E * 128;
  float s = 0.0f;
#pragma unroll
  for (int p = 0; p < SPLITK; ++p) s += m1part[p * stride + idx];
  m1T[(size_t)d * E + e] = to_bf16(s * inv_e[e]);
}

// ---------------------------------------------------------------------------
// 6) GEMM2: m2s = dv^-1/2 * (H @ m1), bf16 out.  A = H rows (fp32 pipelined),
//    B = m1T via async LDS loads.  Double-buffered.
// ---------------------------------------------------------------------------
__global__ __launch_bounds__(256)
void gemm2_kernel(const float* __restrict__ H, const __bf16* __restrict__ m1T,
                  const float* __restrict__ inv_v, __bf16* __restrict__ m2s,
                  int E) {
  __shared__ __bf16 At[2][128 * LDT];
  __shared__ __bf16 Bt[2][128 * LDT];
  const int mb   = blockIdx.x * 128;
  const int tid  = threadIdx.x;
  const int lane = tid & 31;
  const int wave = tid >> 5;
  const int wr   = wave & 3;
  const int wc   = wave >> 2;
  const int nk   = E / KC;

  const int am   = tid >> 1;          // A: row m (0..127)
  const int ahalf = (tid & 1) * 16;   // A: k half
  const int bd   = tid >> 1;
  const int bhalf = (tid & 1) * 16;

  f32x8 zero = {0.f,0.f,0.f,0.f,0.f,0.f,0.f,0.f};
  f32x8 acc[2][4];
#pragma unroll
  for (int i = 0; i < 2; ++i)
#pragma unroll
    for (int j = 0; j < 4; ++j) acc[i][j] = zero;

  float4 f0, f1, f2, f3;

  auto issueA = [&](int kk) {
    const float4* s = (const float4*)(H + (size_t)(mb + am) * E + kk * KC + ahalf);
    f0 = s[0]; f1 = s[1]; f2 = s[2]; f3 = s[3];
  };
  auto storeA = [&](int buf) {        // fused fp32->bf16, row-major
    __bf16 tmp[16] __attribute__((aligned(16)));
    float fv[16] = {f0.x,f0.y,f0.z,f0.w, f1.x,f1.y,f1.z,f1.w,
                    f2.x,f2.y,f2.z,f2.w, f3.x,f3.y,f3.z,f3.w};
#pragma unroll
    for (int i = 0; i < 16; ++i) tmp[i] = to_bf16(fv[i]);
    *(uint4*)&At[buf][am * LDT + ahalf]     = *(const uint4*)&tmp[0];
    *(uint4*)&At[buf][am * LDT + ahalf + 8] = *(const uint4*)&tmp[8];
  };
  auto issueB = [&](int kk, int buf) {
    const __bf16* g = m1T + (size_t)bd * E + kk * KC + bhalf;
    copy_b128_to_lds(&Bt[buf][bd * LDT + bhalf],     g);
    copy_b128_to_lds(&Bt[buf][bd * LDT + bhalf + 8], g + 8);
  };

  issueA(0); issueB(0, 0); storeA(0);
  for (int kk = 0; kk < nk; ++kk) {
    const int cur = kk & 1;
    wait_async0();
    __syncthreads();
    if (kk + 1 < nk) { issueA(kk + 1); issueB(kk + 1, 1 - cur); }

    FragBF16 a[2], b[4];
#pragma unroll
    for (int i = 0; i < 2; ++i) a[i] = load_frag_a(&At[cur][(wr * 32 + i * 16) * LDT], lane);
#pragma unroll
    for (int j = 0; j < 4; ++j) b[j] = load_frag_b(&Bt[cur][(wc * 64 + j * 16) * LDT], lane);
#pragma unroll
    for (int i = 0; i < 2; ++i)
#pragma unroll
      for (int j = 0; j < 4; ++j) acc[i][j] = wmma_bf16(a[i], b[j], acc[i][j]);
    __syncthreads();
    if (kk + 1 < nk) storeA(1 - cur);
  }

  const int rl = (lane >> 4) * 8;
  const int cl = lane & 15;
#pragma unroll
  for (int i = 0; i < 2; ++i) {
    int rbase = mb + wr * 32 + i * 16 + rl;
#pragma unroll
    for (int j = 0; j < 4; ++j) {
      int d = wc * 64 + j * 16 + cl;
#pragma unroll
      for (int v = 0; v < 8; ++v) {
        int row = rbase + v;
        m2s[(size_t)row * 128 + d] = to_bf16(acc[i][j][v] * inv_v[row]);
      }
    }
  }
}

// ---------------------------------------------------------------------------
// 7) GEMM3: h = relu(m2s @ W^T + bias).  A = m2s bf16 via async LDS loads,
//    B = W fp32 pipelined (Bt[j][k] = W[j][k] is already the transposed-B
//    layout).  Double-buffered, nk = 4.
// ---------------------------------------------------------------------------
__global__ __launch_bounds__(256)
void gemm3_kernel(const __bf16* __restrict__ m2s, const float* __restrict__ W,
                  const float* __restrict__ bias, float* __restrict__ h) {
  __shared__ __bf16 At[2][128 * LDT];
  __shared__ __bf16 Bt[2][128 * LDT];
  const int mb   = blockIdx.x * 128;
  const int tid  = threadIdx.x;
  const int lane = tid & 31;
  const int wave = tid >> 5;
  const int wr   = wave & 3;
  const int wc   = wave >> 2;
  const int nk   = 128 / KC;          // 4

  const int am    = tid >> 1;
  const int ahalf = (tid & 1) * 16;

  f32x8 zero = {0.f,0.f,0.f,0.f,0.f,0.f,0.f,0.f};
  f32x8 acc[2][4];
#pragma unroll
  for (int i = 0; i < 2; ++i)
#pragma unroll
    for (int j = 0; j < 4; ++j) acc[i][j] = zero;

  float4 f0, f1, f2, f3;

  auto issueA = [&](int kk, int buf) {
    const __bf16* g = m2s + (size_t)(mb + am) * 128 + kk * KC + ahalf;
    copy_b128_to_lds(&At[buf][am * LDT + ahalf],     g);
    copy_b128_to_lds(&At[buf][am * LDT + ahalf + 8], g + 8);
  };
  auto issueB = [&](int kk) {
    const float4* s = (const float4*)(W + (size_t)am * 128 + kk * KC + ahalf);
    f0 = s[0]; f1 = s[1]; f2 = s[2]; f3 = s[3];
  };
  auto storeB = [&](int buf) {
    __bf16 tmp[16] __attribute__((aligned(16)));
    float fv[16] = {f0.x,f0.y,f0.z,f0.w, f1.x,f1.y,f1.z,f1.w,
                    f2.x,f2.y,f2.z,f2.w, f3.x,f3.y,f3.z,f3.w};
#pragma unroll
    for (int i = 0; i < 16; ++i) tmp[i] = to_bf16(fv[i]);
    *(uint4*)&Bt[buf][am * LDT + ahalf]     = *(const uint4*)&tmp[0];
    *(uint4*)&Bt[buf][am * LDT + ahalf + 8] = *(const uint4*)&tmp[8];
  };

  issueA(0, 0); issueB(0); storeB(0);
  for (int kk = 0; kk < nk; ++kk) {
    const int cur = kk & 1;
    wait_async0();
    __syncthreads();
    if (kk + 1 < nk) { issueA(kk + 1, 1 - cur); issueB(kk + 1); }

    FragBF16 a[2], b[4];
#pragma unroll
    for (int i = 0; i < 2; ++i) a[i] = load_frag_a(&At[cur][(wr * 32 + i * 16) * LDT], lane);
#pragma unroll
    for (int j = 0; j < 4; ++j) b[j] = load_frag_b(&Bt[cur][(wc * 64 + j * 16) * LDT], lane);
#pragma unroll
    for (int i = 0; i < 2; ++i)
#pragma unroll
      for (int j = 0; j < 4; ++j) acc[i][j] = wmma_bf16(a[i], b[j], acc[i][j]);
    __syncthreads();
    if (kk + 1 < nk) storeB(1 - cur);
  }

  const int rl = (lane >> 4) * 8;
  const int cl = lane & 15;
#pragma unroll
  for (int i = 0; i < 2; ++i) {
    int rbase = mb + wr * 32 + i * 16 + rl;
#pragma unroll
    for (int j = 0; j < 4; ++j) {
      int d = wc * 64 + j * 16 + cl;
      float bv = bias[d];
#pragma unroll
      for (int v = 0; v < 8; ++v) {
        float t = acc[i][j][v] + bv;
        h[(size_t)(rbase + v) * 128 + d] = t > 0.0f ? t : 0.0f;
      }
    }
  }
}

// ---------------------------------------------------------------------------
// 8) layernorm: one wave32 per row (128 feats = 4 per lane), shfl reductions
// ---------------------------------------------------------------------------
__global__ __launch_bounds__(256)
void layernorm_kernel(const float* __restrict__ h, const float* __restrict__ gamma,
                      const float* __restrict__ beta, float* __restrict__ out) {
  const int lane = threadIdx.x & 31;
  const int wave = threadIdx.x >> 5;
  const int row  = blockIdx.x * 8 + wave;
  float4 v = ((const float4*)(h + (size_t)row * 128))[lane];
  float s  = v.x + v.y + v.z + v.w;
  float sq = v.x * v.x + v.y * v.y + v.z * v.z + v.w * v.w;
#pragma unroll
  for (int m = 16; m > 0; m >>= 1) {
    s  += __shfl_xor(s,  m, 32);
    sq += __shfl_xor(sq, m, 32);
  }
  float mu  = s * (1.0f / 128.0f);
  float var = sq * (1.0f / 128.0f) - mu * mu;
  float rs  = rsqrtf(var + 1e-5f);
  float4 g  = ((const float4*)gamma)[lane];
  float4 be = ((const float4*)beta)[lane];
  float4 o;
  o.x = g.x * (v.x - mu) * rs + be.x;
  o.y = g.y * (v.y - mu) * rs + be.y;
  o.z = g.z * (v.z - mu) * rs + be.z;
  o.w = g.w * (v.w - mu) * rs + be.w;
  ((float4*)(out + (size_t)row * 128))[lane] = o;
}

// ---------------------------------------------------------------------------
// launch
// ---------------------------------------------------------------------------
extern "C" void kernel_launch(void* const* d_in, const int* in_sizes, int n_in,
                              void* d_out, int out_size, void* d_ws, size_t ws_size,
                              hipStream_t stream) {
  const float* x     = (const float*)d_in[0];
  const float* Hm    = (const float*)d_in[1];
  const float* W     = (const float*)d_in[2];
  const float* bias  = (const float*)d_in[3];
  const float* gamma = (const float*)d_in[4];
  const float* beta  = (const float*)d_in[5];
  float* out = (float*)d_out;

  const int D = 128;
  const int N = in_sizes[0] / D;        // 32768
  const int E = in_sizes[1] / N;        // 8192

  // workspace carve-out (256B aligned)
  char* ws = (char*)d_ws;
  size_t off = 0;
  auto carve = [&](size_t bytes) -> char* {
    char* p = ws + off;
    off = (off + bytes + 255) & ~(size_t)255;
    return p;
  };
  float*  d_v    = (float*)carve((size_t)N * 4);
  float*  d_e    = (float*)carve((size_t)E * 4);
  float*  inv_v  = (float*)carve((size_t)N * 4);
  float*  inv_e  = (float*)carve((size_t)E * 4);
  __bf16* xsT    = (__bf16*)carve((size_t)128 * N * 2);
  float*  m1part = (float*)carve((size_t)SPLITK * E * 128 * 4);
  __bf16* m1T    = (__bf16*)carve((size_t)128 * E * 2);
  __bf16* m2s    = (__bf16*)carve((size_t)N * 128 * 2);
  float*  hbuf   = (float*)carve((size_t)N * 128 * 4);
  (void)ws_size; (void)n_in; (void)out_size;

  zero_f32_kernel<<<(E + 255) / 256, 256, 0, stream>>>(d_e, E);
  degrees_kernel<<<N / 32, 256, 0, stream>>>(Hm, d_v, d_e, E);
  int mx = N > E ? N : E;
  inv_deg_kernel<<<(mx + 255) / 256, 256, 0, stream>>>(d_v, d_e, inv_v, inv_e, N, E);
  scale_x_kernel<<<N / 64, 256, 0, stream>>>(x, inv_v, xsT, N);
  gemm1_kernel<<<dim3(E / 128, SPLITK), 256, 0, stream>>>(Hm, xsT, m1part, N, E, N / SPLITK);
  reduce_m1_kernel<<<(E * 128) / 256, 256, 0, stream>>>(m1part, inv_e, m1T, E);
  gemm2_kernel<<<N / 128, 256, 0, stream>>>(Hm, m1T, inv_v, m2s, E);
  gemm3_kernel<<<N / 128, 256, 0, stream>>>(m2s, W, bias, hbuf);
  layernorm_kernel<<<N / 8, 256, 0, stream>>>(hbuf, gamma, beta, out);
}
